// Pose_and_View_SwarmNet_32753420599778
// MI455X (gfx1250) — matrix operand order repaired and checked by reading, not compiled, not run on hardware
//
#include <hip/hip_runtime.h>

typedef __attribute__((ext_vector_type(16))) _Float16 v16h;
typedef __attribute__((ext_vector_type(8)))  float    v8f;
typedef __attribute__((ext_vector_type(4)))  int      v4i;

#define N_NODES 50000
#define N_EDGES 4950000
#define HID     256
#define KG      64            // GraphConv K (40) padded to 64 for two k=32 WMMA steps
#define LRELU(x) ((x) >= 0.f ? (x) : 0.1f * (x))

// Async global->LDS path (gfx1250): guarded so compile never breaks.
#if defined(__has_builtin)
#  if __has_builtin(__builtin_amdgcn_global_load_async_to_lds_b128) && \
      __has_builtin(__builtin_amdgcn_s_wait_asynccnt)
#    define ASYNC_COPY 1
#  endif
#endif
#ifndef ASYNC_COPY
#  define ASYNC_COPY 0
#endif

typedef __attribute__((address_space(1))) v4i* gv4i_p;   // global int4*
typedef __attribute__((address_space(3))) v4i* sv4i_p;   // LDS    int4*

union HFrag { v16h v; float4 q[2]; };
union CFrag { v8f  v; float  f[8]; };

// ---- WMMA fragment loaders (layouts per CDNA5 ISA 7.12.2, wave32) ----------
__device__ __forceinline__ void load_b_frag(HFrag& b, const _Float16* __restrict__ W,
                                            int col, int ldb, int k0, int hi) {
  const _Float16* p = W + (size_t)col * ldb + k0 + 8 * hi;
  b.q[0] = *(const float4*)(p);        // halves 0..7  : K = k0+8*hi    .. +7
  b.q[1] = *(const float4*)(p + 16);   // halves 8..15 : K = k0+16+8*hi .. +7
}
__device__ __forceinline__ void load_a_frag_g(HFrag& a, const _Float16* __restrict__ A,
                                              int row, int lda, int k0, int hi) {
  const _Float16* p = A + (size_t)row * lda + k0 + 8 * hi;
  a.q[0] = *(const float4*)(p);        // halves 0..7  : K = k0+8*hi    .. +7
  a.q[1] = *(const float4*)(p + 16);   // halves 8..15 : K = k0+16+8*hi .. +7
}

// ---- view encoder: conv1d(3->9,k3,s2) -> lrelu -> conv1d(9->1,k3,s2) -> lrelu
__global__ __launch_bounds__(256) void encoder_kernel(
    const float* __restrict__ pose, const float* __restrict__ views,
    const float* __restrict__ w1, const float* __restrict__ b1,
    const float* __restrict__ w2, const float* __restrict__ b2,
    float* __restrict__ x32, _Float16* __restrict__ xh)
{
  int n = blockIdx.x * blockDim.x + threadIdx.x;
  if (n >= N_NODES) return;
  const float* vw = views + (size_t)n * 3 * 151;
  float*       xo  = x32 + (size_t)n * 40;
  _Float16*    xho = xh  + (size_t)n * KG;
  #pragma unroll
  for (int c = 0; c < 3; ++c) { float p = pose[(size_t)n * 3 + c]; xo[c] = p; xho[c] = (_Float16)p; }

  float c0[9], c1[9], c2[9];
  auto mkcol = [&](int p, float* dst) {
    #pragma unroll
    for (int oc = 0; oc < 9; ++oc) {
      float s = b1[oc];
      #pragma unroll
      for (int ic = 0; ic < 3; ++ic)
        #pragma unroll
        for (int k = 0; k < 3; ++k)
          s += w1[oc * 9 + ic * 3 + k] * vw[ic * 151 + 2 * p + k];
      dst[oc] = LRELU(s);
    }
  };
  mkcol(0, c0); mkcol(1, c1); mkcol(2, c2);
  for (int j = 0;;) {
    float s = b2[0];
    #pragma unroll
    for (int oc = 0; oc < 9; ++oc)
      s += w2[oc * 3 + 0] * c0[oc] + w2[oc * 3 + 1] * c1[oc] + w2[oc * 3 + 2] * c2[oc];
    float v = LRELU(s);
    xo[3 + j] = v; xho[3 + j] = (_Float16)v;
    if (++j == 37) break;
    #pragma unroll
    for (int oc = 0; oc < 9; ++oc) c0[oc] = c2[oc];
    mkcol(2 * j + 1, c1); mkcol(2 * j + 2, c2);
  }
  #pragma unroll
  for (int c = 40; c < KG; ++c) xho[c] = (_Float16)0.f;   // zero K padding
}

__global__ __launch_bounds__(256) void zero_f32_kernel(float* __restrict__ p, int n) {
  int i = blockIdx.x * blockDim.x + threadIdx.x;
  if (i < n) p[i] = 0.f;
}

// f32 [rows,scols] -> f16 [rows,dcols] zero-padded
__global__ __launch_bounds__(256) void cvt_pad_kernel(const float* __restrict__ src,
                                                      _Float16* __restrict__ dst,
                                                      int rows, int scols, int dcols) {
  int i = blockIdx.x * blockDim.x + threadIdx.x;
  if (i >= rows * dcols) return;
  int r = i / dcols, c = i - r * dcols;
  dst[i] = (c < scols) ? (_Float16)src[(size_t)r * scols + c] : (_Float16)0.f;
}

// edge scatter: agg[dst] += x[src]; 8 threads per edge, 5 feats each (coalesced gather)
__global__ __launch_bounds__(256) void scatter_add_kernel(const int* __restrict__ ei,
                                                          const float* __restrict__ x,
                                                          float* __restrict__ agg) {
  long long t = (long long)blockIdx.x * blockDim.x + threadIdx.x;
  if (t >= (long long)N_EDGES * 8) return;
  int e  = (int)(t >> 3);
  int c0 = (int)(t & 7);
  int s = ei[e], d = ei[N_EDGES + e];
  const float* xs = x + (size_t)s * 40;
  float*       ad = agg + (size_t)d * 40;
  #pragma unroll
  for (int i = 0; i < 5; ++i) {
    int c = c0 + 8 * i;
    atomicAdd(ad + c, xs[c]);
  }
}

// h = lrelu(aggH @ wrelH^T + b_rel + xH @ wrootH^T), one wave per 16x16 tile
__global__ __launch_bounds__(256) void graphconv_kernel(
    const _Float16* __restrict__ aggh, const _Float16* __restrict__ xh,
    const _Float16* __restrict__ wrel, const _Float16* __restrict__ wroot,
    const float* __restrict__ brel, _Float16* __restrict__ out)
{
  int wave = threadIdx.x >> 5, lane = threadIdx.x & 31;
  int tile = blockIdx.x * 8 + wave;
  const int colTiles = HID / 16;
  int rt = tile / colTiles, ct = tile - rt * colTiles;
  if (rt * 16 >= N_NODES) return;
  int m = lane & 15, hi = lane >> 4;
  int row = rt * 16 + m, col = ct * 16 + m;

  CFrag acc = {};
  HFrag a, b;
  #pragma unroll
  for (int k0 = 0; k0 < KG; k0 += 32) {
    load_a_frag_g(a, aggh, row, KG, k0, hi);
    load_b_frag(b, wrel, col, KG, k0, hi);
    acc.v = __builtin_amdgcn_wmma_f32_16x16x32_f16(false, a.v, false, b.v,
                                                   (short)0, acc.v, false, false);
  }
  #pragma unroll
  for (int k0 = 0; k0 < KG; k0 += 32) {
    load_a_frag_g(a, xh, row, KG, k0, hi);
    load_b_frag(b, wroot, col, KG, k0, hi);
    acc.v = __builtin_amdgcn_wmma_f32_16x16x32_f16(false, a.v, false, b.v,
                                                   (short)0, acc.v, false, false);
  }
  float bv = brel[col];
  #pragma unroll
  for (int r = 0; r < 8; ++r) {
    float v = acc.f[r] + bv;
    v = LRELU(v);
    out[(size_t)(rt * 16 + r + 8 * hi) * HID + col] = (_Float16)v;
  }
}

// out = lrelu(A[N,256] @ W[256,256]^T + bias)
// block = 8 waves = one 16-row stripe; A stripe staged once into LDS (async copy),
// each wave computes 2 column tiles sharing the LDS A fragments.
__global__ __launch_bounds__(256) void mlp_kernel(
    const _Float16* __restrict__ A, const _Float16* __restrict__ W,
    const float* __restrict__ bias, _Float16* __restrict__ out)
{
  __shared__ __attribute__((aligned(16))) _Float16 sA[16 * HID];   // 8 KB stripe
  const int rt = blockIdx.x;
  const int t  = threadIdx.x;

  // stage rows [rt*16, rt*16+16) of A: 8 KB contiguous, 2x 16B chunks per thread
  const _Float16* gbase = A + (size_t)rt * 16 * HID;
  #pragma unroll
  for (int i = 0; i < 2; ++i) {
    int off = (t + i * 256) * 8;     // offset in halves (16B granularity)
#if ASYNC_COPY
    __builtin_amdgcn_global_load_async_to_lds_b128(
        (gv4i_p)(v4i*)(gbase + off),
        (sv4i_p)(v4i*)(sA + off), 0, 0);
#else
    *(float4*)(sA + off) = *(const float4*)(gbase + off);
#endif
  }
#if ASYNC_COPY
  __builtin_amdgcn_s_wait_asynccnt(0);
#endif
  __syncthreads();

  const int wave = t >> 5, lane = t & 31;
  const int m = lane & 15, hi = lane >> 4;
  const int col0 = wave * 16 + m;
  const int col1 = (wave + 8) * 16 + m;

  CFrag acc0 = {}, acc1 = {};
  HFrag a, b0, b1;
  #pragma unroll
  for (int k0 = 0; k0 < HID; k0 += 32) {
    const _Float16* ap = sA + m * HID + k0 + 8 * hi;   // ds_load_b128 x2
    a.q[0] = *(const float4*)(ap);
    a.q[1] = *(const float4*)(ap + 16);
    load_b_frag(b0, W, col0, HID, k0, hi);
    load_b_frag(b1, W, col1, HID, k0, hi);
    acc0.v = __builtin_amdgcn_wmma_f32_16x16x32_f16(false, a.v, false, b0.v,
                                                    (short)0, acc0.v, false, false);
    acc1.v = __builtin_amdgcn_wmma_f32_16x16x32_f16(false, a.v, false, b1.v,
                                                    (short)0, acc1.v, false, false);
  }
  float bv0 = bias[col0], bv1 = bias[col1];
  #pragma unroll
  for (int r = 0; r < 8; ++r) {
    float v0 = acc0.f[r] + bv0; v0 = LRELU(v0);
    float v1 = acc1.f[r] + bv1; v1 = LRELU(v1);
    size_t rowi = (size_t)(rt * 16 + r + 8 * hi) * HID;
    out[rowi + col0] = (_Float16)v0;
    out[rowi + col1] = (_Float16)v1;
  }
}

// out[n] = h3[n,:] . w_pred + b_pred   (256-dot, f32 accumulate)
__global__ __launch_bounds__(256) void pred_kernel(const _Float16* __restrict__ h,
                                                   const float* __restrict__ wp,
                                                   const float* __restrict__ bp,
                                                   float* __restrict__ out) {
  int n = blockIdx.x * blockDim.x + threadIdx.x;
  if (n >= N_NODES) return;
  const _Float16* hp = h + (size_t)n * HID;
  float s = bp[0];
  #pragma unroll 8
  for (int c = 0; c < HID; ++c) s += (float)hp[c] * wp[c];
  out[n] = s;
}

extern "C" void kernel_launch(void* const* d_in, const int* in_sizes, int n_in,
                              void* d_out, int out_size, void* d_ws, size_t ws_size,
                              hipStream_t stream) {
  (void)in_sizes; (void)n_in; (void)out_size; (void)ws_size;
  const float* pose   = (const float*)d_in[0];
  const float* views  = (const float*)d_in[1];
  const int*   eidx   = (const int*)d_in[2];
  const float* w_e1   = (const float*)d_in[3];
  const float* b_e1   = (const float*)d_in[4];
  const float* w_e2   = (const float*)d_in[5];
  const float* b_e2   = (const float*)d_in[6];
  const float* w_rel  = (const float*)d_in[7];
  const float* b_rel  = (const float*)d_in[8];
  const float* w_root = (const float*)d_in[9];
  const float* w_l1   = (const float*)d_in[10];
  const float* b_l1   = (const float*)d_in[11];
  const float* w_l2   = (const float*)d_in[12];
  const float* b_l2   = (const float*)d_in[13];
  const float* w_l3   = (const float*)d_in[14];
  const float* b_l3   = (const float*)d_in[15];
  const float* w_pred = (const float*)d_in[16];
  const float* b_pred = (const float*)d_in[17];
  float* out = (float*)d_out;

  // workspace carve (256B-aligned)
  size_t off = 0;
  auto carve = [&](size_t bytes) -> void* {
    void* p = (char*)d_ws + off;
    off += (bytes + 255) & ~(size_t)255;
    return p;
  };
  float*    x32   = (float*)   carve((size_t)N_NODES * 40 * sizeof(float));
  float*    agg32 = (float*)   carve((size_t)N_NODES * 40 * sizeof(float));
  _Float16* xh    = (_Float16*)carve((size_t)N_NODES * KG * sizeof(_Float16));
  _Float16* aggh  = (_Float16*)carve((size_t)N_NODES * KG * sizeof(_Float16));
  _Float16* hA    = (_Float16*)carve((size_t)N_NODES * HID * sizeof(_Float16));
  _Float16* hB    = (_Float16*)carve((size_t)N_NODES * HID * sizeof(_Float16));
  _Float16* wrelh  = (_Float16*)carve((size_t)HID * KG * sizeof(_Float16));
  _Float16* wrooth = (_Float16*)carve((size_t)HID * KG * sizeof(_Float16));
  _Float16* wl1h   = (_Float16*)carve((size_t)HID * HID * sizeof(_Float16));
  _Float16* wl2h   = (_Float16*)carve((size_t)HID * HID * sizeof(_Float16));
  _Float16* wl3h   = (_Float16*)carve((size_t)HID * HID * sizeof(_Float16));

  const int B = 256;
  auto gridFor = [](long long n, int b) { return (int)((n + b - 1) / b); };

  // 1) zero the fp32 aggregation buffer
  zero_f32_kernel<<<gridFor(N_NODES * 40, B), B, 0, stream>>>(agg32, N_NODES * 40);
  // 2) view encoder + concat -> x (fp32 for scatter, f16 padded for WMMA)
  encoder_kernel<<<gridFor(N_NODES, B), B, 0, stream>>>(pose, views, w_e1, b_e1, w_e2, b_e2, x32, xh);
  // 3) weight conversions to f16 (K padded to 64 for GraphConv weights)
  cvt_pad_kernel<<<gridFor(HID * KG, B), B, 0, stream>>>(w_rel,  wrelh,  HID, 40, KG);
  cvt_pad_kernel<<<gridFor(HID * KG, B), B, 0, stream>>>(w_root, wrooth, HID, 40, KG);
  cvt_pad_kernel<<<gridFor(HID * HID, B), B, 0, stream>>>(w_l1, wl1h, HID, HID, HID);
  cvt_pad_kernel<<<gridFor(HID * HID, B), B, 0, stream>>>(w_l2, wl2h, HID, HID, HID);
  cvt_pad_kernel<<<gridFor(HID * HID, B), B, 0, stream>>>(w_l3, wl3h, HID, HID, HID);
  // 4) edge scatter-add (fp32 atomics into L2-resident 8 MB buffer)
  scatter_add_kernel<<<gridFor((long long)N_EDGES * 8, B), B, 0, stream>>>(eidx, x32, agg32);
  // 5) agg -> f16 padded
  cvt_pad_kernel<<<gridFor(N_NODES * KG, B), B, 0, stream>>>(agg32, aggh, N_NODES, 40, KG);

  // 6) GraphConv (one wave per tile) + MLP stack (LDS-staged A stripes)
  const int tiles  = (N_NODES / 16) * (HID / 16);   // 50000
  const int gcBlocks = tiles / 8;                   // 6250
  const int rowBlocks = N_NODES / 16;               // 3125
  graphconv_kernel<<<gcBlocks, B, 0, stream>>>(aggh, xh, wrelh, wrooth, b_rel, hA);
  mlp_kernel<<<rowBlocks, B, 0, stream>>>(hA, wl1h, b_l1, hB);
  mlp_kernel<<<rowBlocks, B, 0, stream>>>(hB, wl2h, b_l2, hA);
  mlp_kernel<<<rowBlocks, B, 0, stream>>>(hA, wl3h, b_l3, hB);
  // 7) final 256 -> 1 projection
  pred_kernel<<<gridFor(N_NODES, B), B, 0, stream>>>(hB, w_pred, b_pred, out);
}
